// LearnedEdge_27762668601697
// MI455X (gfx1250) — matrix operand order repaired and checked by reading, not compiled, not run on hardware
//
#include <hip/hip_runtime.h>
#include <hip/hip_bf16.h>
#include <hip/hip_fp16.h>

typedef __attribute__((ext_vector_type(16))) _Float16 v16h;
typedef __attribute__((ext_vector_type(8)))  float    v8f;

#define Bn   8
#define Nn   384
#define Fn   128
#define Sn   5
#define EPB  73536          /* N*(N-1)/2 */
#define ETOT (Bn*EPB)       /* 588288 */
#define TILES (ETOT/16)     /* 36768 */
#define NEGF (-3.402823466e38f)

/* ws layout (bytes): w1f and w2f contiguous so one async stage covers both */
#define W1F_OFF 0           /* 32768 halves = 65536 B */
#define W2F_OFF 65536       /* 16384 halves = 32768 B */
#define WF_BYTES 98304      /* total fragment bytes staged to LDS */
#define LOG_OFF 98304       /* 588288 floats */

/* ---------------- prep: swizzle W1/W2 into WMMA B-fragment layout, f32->f16 ------------- */
__global__ __launch_bounds__(256) void prep_kernel(const float* __restrict__ W1,
                                                   const float* __restrict__ W2,
                                                   _Float16* __restrict__ w1f,
                                                   _Float16* __restrict__ w2f) {
    int idx = blockIdx.x * 256 + threadIdx.x;
    if (idx < 32768) {
        int t    = idx & 15;
        int lane = (idx >> 4) & 31;
        int nt   = (idx >> 9) & 7;
        int ks   = idx >> 12;                 /* 0..7 */
        int k = 32 * ks + (lane >> 4) * 16 + t;
        int n = nt * 16 + (lane & 15);
        w1f[idx] = (_Float16)W1[k * Fn + n];
    } else if (idx < 49152) {
        int i2   = idx - 32768;
        int t    = i2 & 15;
        int lane = (i2 >> 4) & 31;
        int nt   = (i2 >> 9) & 7;
        int ks   = i2 >> 12;                  /* 0..3 */
        int k = 32 * ks + (lane >> 4) * 16 + t;
        int n = nt * 16 + (lane & 15);
        w2f[i2] = (_Float16)W2[k * Fn + n];
    }
}

/* ---------------- edge MLP: WMMA GEMMs + fused bias/ReLU/LayerNorm ---------------------- */
__global__ __launch_bounds__(256) void edge_mlp_kernel(
    const float* __restrict__ nodes,
    const float* __restrict__ b1, const float* __restrict__ g1, const float* __restrict__ bt1,
    const float* __restrict__ b2, const float* __restrict__ g2, const float* __restrict__ bt2,
    const float* __restrict__ W3, const float* __restrict__ b3,
    const int* __restrict__ batch_idx, const int* __restrict__ sink_idx,
    const int* __restrict__ source_idx,
    const _Float16* __restrict__ w1f,   /* contiguous w1f|w2f fragment block */
    float* __restrict__ logits)
{
    __shared__ _Float16 lds_h1[8][16 * 136];   /* per-wave 16x128 tile, padded stride */
    extern __shared__ _Float16 wsh[];          /* 49152 halves: w1 frags | w2 frags */

    const int tid  = threadIdx.x;

    /* ---- stage weight fragments to LDS with CDNA5 async copies (ASYNCcnt) ---- */
    {
        unsigned ldsbase = (unsigned)(uintptr_t)(&wsh[0]);   /* WG-relative LDS byte offset */
        unsigned long long gbase = (unsigned long long)(uintptr_t)w1f;
        #pragma unroll
        for (int i = 0; i < 24; ++i) {
            unsigned off = (unsigned)(tid + i * 256) * 16u;  /* 16B per lane per round */
            asm volatile("global_load_async_to_lds_b128 %0, %1, %2"
                         :: "v"(ldsbase + off), "v"(off), "s"(gbase)
                         : "memory");
        }
        asm volatile("s_wait_asynccnt 0x0" ::: "memory");
    }
    __syncthreads();

    const _Float16* w1s = &wsh[0];
    const _Float16* w2s = &wsh[32768];

    const int w    = tid >> 5;          /* wave in block */
    const int l    = tid & 31;          /* lane */
    const int ln   = l & 15;
    const int hw   = l >> 4;            /* half-wave */
    const int koff = hw * 8;
    const int tile = blockIdx.x * 8 + w;
    const int e    = tile * 16 + ln;    /* edge row handled by this lane (both halves) */

    const int bidx = batch_idx[e];
    const float* sinkp = nodes + ((size_t)bidx * Nn + sink_idx[e]) * Fn;
    const float* srcp  = nodes + ((size_t)bidx * Nn + source_idx[e]) * Fn;

    float bias1[8], gam1[8], bet1[8], bias2[8], gam2[8], bet2[8], w3v[8];
    #pragma unroll
    for (int nt = 0; nt < 8; ++nt) {
        int n = nt * 16 + ln;
        bias1[nt] = b1[n]; gam1[nt] = g1[n]; bet1[nt] = bt1[n];
        bias2[nt] = b2[n]; gam2[nt] = g2[n]; bet2[nt] = bt2[n];
        w3v[nt]   = W3[n];
    }

    v8f zero = {};
    v8f acc[8];
    #pragma unroll
    for (int nt = 0; nt < 8; ++nt) acc[nt] = zero;

    /* ---- GEMM1: X[16,256] @ W1[256,128], K split sink(ks 0..3)/source(ks 4..7) ---- */
    for (int ks = 0; ks < 8; ++ks) {
        const float* base = (ks < 4) ? (sinkp + 32 * ks) : (srcp + 32 * (ks - 4));
        v16h a;
        #pragma unroll
        for (int r = 0; r < 8; ++r) {
            int kk = hw * 16 + 2 * r;
            float2 f2 = *(const float2*)(base + kk);
            a[2 * r]     = (_Float16)f2.x;
            a[2 * r + 1] = (_Float16)f2.y;
        }
        const _Float16* wp = w1s + (size_t)(ks * 8) * 32 * 16;
        #pragma unroll
        for (int nt = 0; nt < 8; ++nt) {
            v16h bf = *(const v16h*)(wp + (nt * 32 + l) * 16);
            acc[nt] = __builtin_amdgcn_wmma_f32_16x16x32_f16(
                false, a, false, bf, (short)0, acc[nt], false, false);
        }
    }

    /* ---- epilogue 1: bias + ReLU + LayerNorm, emit f16 H1 into LDS ---- */
    float vals[8][8];
    #pragma unroll
    for (int nt = 0; nt < 8; ++nt)
        #pragma unroll
        for (int j = 0; j < 8; ++j) {
            float v = acc[nt][j] + bias1[nt];
            vals[nt][j] = v > 0.f ? v : 0.f;
        }

    _Float16* myLds = &lds_h1[w][0];
    #pragma unroll
    for (int j = 0; j < 8; ++j) {
        float s1 = 0.f, s2 = 0.f;
        #pragma unroll
        for (int nt = 0; nt < 8; ++nt) { float v = vals[nt][j]; s1 += v; s2 += v * v; }
        #pragma unroll
        for (int off = 1; off < 16; off <<= 1) {
            s1 += __shfl_xor(s1, off, 32);
            s2 += __shfl_xor(s2, off, 32);
        }
        float mean = s1 * (1.f / 128.f);
        float var  = s2 * (1.f / 128.f) - mean * mean;
        float inv  = rsqrtf(var + 1e-5f);
        int m = j + 8 * hw;
        #pragma unroll
        for (int nt = 0; nt < 8; ++nt) {
            float h = (vals[nt][j] - mean) * inv * gam1[nt] + bet1[nt];
            myLds[m * 136 + nt * 16 + ln] = (_Float16)h;
        }
    }
    /* intra-wave cross-lane LDS dependency: wave-synchronous, just drain DS */
    asm volatile("s_wait_dscnt 0" ::: "memory");

    /* ---- GEMM2: H1[16,128] @ W2[128,128] ---- */
    v8f acc2[8];
    #pragma unroll
    for (int nt = 0; nt < 8; ++nt) acc2[nt] = zero;

    for (int ks = 0; ks < 4; ++ks) {
        v16h a;
        #pragma unroll
        for (int r = 0; r < 8; ++r) {
            int kk = 32 * ks + hw * 16 + 2 * r;
            a[2 * r]     = myLds[ln * 136 + kk];
            a[2 * r + 1] = myLds[ln * 136 + kk + 1];
        }
        const _Float16* wp = w2s + (size_t)(ks * 8) * 32 * 16;
        #pragma unroll
        for (int nt = 0; nt < 8; ++nt) {
            v16h bf = *(const v16h*)(wp + (nt * 32 + l) * 16);
            acc2[nt] = __builtin_amdgcn_wmma_f32_16x16x32_f16(
                false, a, false, bf, (short)0, acc2[nt], false, false);
        }
    }

    /* ---- epilogue 2: bias + ReLU + LN + dot with W3 ---- */
    #pragma unroll
    for (int nt = 0; nt < 8; ++nt)
        #pragma unroll
        for (int j = 0; j < 8; ++j) {
            float v = acc2[nt][j] + bias2[nt];
            vals[nt][j] = v > 0.f ? v : 0.f;
        }

    float myLogit = 0.f;
    #pragma unroll
    for (int j = 0; j < 8; ++j) {
        float s1 = 0.f, s2 = 0.f;
        #pragma unroll
        for (int nt = 0; nt < 8; ++nt) { float v = vals[nt][j]; s1 += v; s2 += v * v; }
        #pragma unroll
        for (int off = 1; off < 16; off <<= 1) {
            s1 += __shfl_xor(s1, off, 32);
            s2 += __shfl_xor(s2, off, 32);
        }
        float mean = s1 * (1.f / 128.f);
        float var  = s2 * (1.f / 128.f) - mean * mean;
        float inv  = rsqrtf(var + 1e-5f);
        float t = 0.f;
        #pragma unroll
        for (int nt = 0; nt < 8; ++nt) {
            float h = (vals[nt][j] - mean) * inv * gam2[nt] + bet2[nt];
            t += h * w3v[nt];
        }
        #pragma unroll
        for (int off = 1; off < 16; off <<= 1) t += __shfl_xor(t, off, 32);
        if (ln == j) myLogit = t;
    }
    if (ln < 8) logits[tile * 16 + ln + 8 * hw] = myLogit + b3[0];
}

/* ---------------- JAX threefry2x32 gumbel + hard gumbel-softmax scatter ----------------- */
__device__ __forceinline__ unsigned rotl32(unsigned x, int r) {
    return (x << r) | (x >> (32 - r));
}

__device__ __forceinline__ float gumbel_at(unsigned i) {
    const unsigned halfn = 2941440u;      /* (5*8*384*383)/2 */
    const unsigned k0 = 0u, k1 = 42u;     /* jax.random.key(42) */
    bool lo = i < halfn;
    unsigned p  = lo ? i : (i - halfn);
    unsigned x0 = p, x1 = p + halfn;
    unsigned ks0 = k0, ks1 = k1, ks2 = 0x1BD11BDAu ^ k0 ^ k1;
    x0 += ks0; x1 += ks1;
#define TF_R(R) { x0 += x1; x1 = rotl32(x1, R); x1 ^= x0; }
    TF_R(13) TF_R(15) TF_R(26) TF_R(6)   x0 += ks1; x1 += ks2 + 1u;
    TF_R(17) TF_R(29) TF_R(16) TF_R(24)  x0 += ks2; x1 += ks0 + 2u;
    TF_R(13) TF_R(15) TF_R(26) TF_R(6)   x0 += ks0; x1 += ks1 + 3u;
    TF_R(17) TF_R(29) TF_R(16) TF_R(24)  x0 += ks1; x1 += ks2 + 4u;
    TF_R(13) TF_R(15) TF_R(26) TF_R(6)   x0 += ks2; x1 += ks0 + 5u;
#undef TF_R
    unsigned bits = lo ? x0 : x1;
    float f = __uint_as_float(0x3F800000u | (bits >> 9)) - 1.0f;
    const float tiny = 1.1754943508222875e-38f;
    float u = f * (1.0f - tiny) + tiny;
    return -logf(-logf(u));
}

__global__ __launch_bounds__(256) void sample_kernel(const float* __restrict__ logits,
                                                     float* __restrict__ adj) {
    __shared__ unsigned long long smax[Sn];
    const int n   = blockIdx.x % Nn;
    const int b   = blockIdx.x / Nn;
    const int tid = threadIdx.x;
    if (tid < Sn) smax[tid] = 0ull;
    __syncthreads();

    unsigned long long lk[Sn];
    #pragma unroll
    for (int s = 0; s < Sn; ++s) lk[s] = 0ull;

    for (int j = tid; j < Nn - 1; j += 256) {
        float L = (j < n) ? logits[b * EPB + (n * (n - 1)) / 2 + j] : NEGF;
        #pragma unroll
        for (int s = 0; s < Sn; ++s) {
            unsigned i = ((unsigned)(s * Bn + b) * Nn + (unsigned)n) * (Nn - 1) + (unsigned)j;
            float v = L + gumbel_at(i);
            unsigned ub  = __float_as_uint(v);
            unsigned ord = (ub & 0x80000000u) ? ~ub : (ub | 0x80000000u);
            /* max val, tie -> min index (matches jnp.argmax first occurrence) */
            unsigned long long key =
                ((unsigned long long)ord << 32) | (unsigned long long)(0xFFFFFFFFu - (unsigned)j);
            if (key > lk[s]) lk[s] = key;
        }
    }
    #pragma unroll
    for (int s = 0; s < Sn; ++s) atomicMax(&smax[s], lk[s]);

    float* row = adj + ((size_t)b * Nn + n) * Nn;
    for (int c = tid; c < Nn; c += 256) row[c] = 0.f;   /* zero row incl. pad col */
    __syncthreads();

    if (tid < Sn) {
        unsigned jj = 0xFFFFFFFFu - (unsigned)(smax[tid] & 0xFFFFFFFFull);
        if ((int)jj != n) row[jj] = 1.0f;               /* diagonal mask */
    }
}

extern "C" void kernel_launch(void* const* d_in, const int* in_sizes, int n_in,
                              void* d_out, int out_size, void* d_ws, size_t ws_size,
                              hipStream_t stream) {
    const float* nodes = (const float*)d_in[0];
    const float* W1    = (const float*)d_in[1];
    const float* b1    = (const float*)d_in[2];
    const float* g1    = (const float*)d_in[3];
    const float* bt1   = (const float*)d_in[4];
    const float* W2    = (const float*)d_in[5];
    const float* b2    = (const float*)d_in[6];
    const float* g2    = (const float*)d_in[7];
    const float* bt2   = (const float*)d_in[8];
    const float* W3    = (const float*)d_in[9];
    const float* b3    = (const float*)d_in[10];
    const int* batch_idx  = (const int*)d_in[11];
    const int* sink_idx   = (const int*)d_in[12];
    const int* source_idx = (const int*)d_in[13];

    char* ws = (char*)d_ws;
    _Float16* w1f    = (_Float16*)(ws + W1F_OFF);
    _Float16* w2f    = (_Float16*)(ws + W2F_OFF);
    float*    logits = (float*)(ws + LOG_OFF);

    prep_kernel<<<192, 256, 0, stream>>>(W1, W2, w1f, w2f);

    edge_mlp_kernel<<<TILES / 8, 256, WF_BYTES, stream>>>(
        nodes, b1, g1, bt1, b2, g2, bt2, W3, b3,
        batch_idx, sink_idx, source_idx, w1f, logits);

    sample_kernel<<<Bn * Nn, 256, 0, stream>>>(logits, (float*)d_out);
}
